// Net_85744727098008
// MI455X (gfx1250) — matrix operand order repaired and checked by reading, not compiled, not run on hardware
//
#include <hip/hip_runtime.h>
#include <stdint.h>

// Problem constants
#define BB   32
#define SS   128
#define TT   64
#define EE   128
#define HH   256
#define VV   32000
#define NWG  16          // LSTM persistent workgroups (gate-column partition)
#define KTOT 384         // E + H
#define KC12 12          // KTOT / 32

typedef __attribute__((ext_vector_type(16))) __bf16        v16bf;
typedef __attribute__((ext_vector_type(8)))  float          v8f;
typedef __attribute__((ext_vector_type(16))) unsigned short v16u;
typedef __attribute__((ext_vector_type(4)))  unsigned int   u32x4;
typedef __attribute__((ext_vector_type(8)))  int            i32x8;
typedef __attribute__((ext_vector_type(4)))  int            i32x4;

union Frag {
  v16u  u;
  v16bf b;
  uint4 q[2];
};

__device__ __forceinline__ unsigned short f2bf(float f) {
  unsigned u = __float_as_uint(f);
  unsigned r = (u + 0x7FFFu + ((u >> 16) & 1u)) >> 16;
  return (unsigned short)r;
}

__device__ __forceinline__ float sigmoidf_(float x) {
  return 1.f / (1.f + __expf(-x));
}
__device__ __forceinline__ float tanhf_(float x) {
  x = fminf(fmaxf(x, -15.f), 15.f);
  float e = __expf(-2.f * x);
  return (1.f - e) / (1.f + e);
}

__device__ __forceinline__ v8f wmma_bf16(v16bf a, v16bf b, v8f c) {
#ifdef __HIP_DEVICE_COMPILE__
  return __builtin_amdgcn_wmma_f32_16x16x32_bf16(false, a, false, b, (short)0, c,
                                                 false, false);
#else
  return c;
#endif
}

// -------------------- agent-scope grid barrier (persistent kernel) ---------
__device__ __forceinline__ void grid_barrier(unsigned* bar) {
  __syncthreads();
  if (threadIdx.x == 0) {
    unsigned gen = __hip_atomic_load(&bar[1], __ATOMIC_RELAXED, __HIP_MEMORY_SCOPE_AGENT);
    __threadfence();
    unsigned old = __hip_atomic_fetch_add(&bar[0], 1u, __ATOMIC_ACQ_REL, __HIP_MEMORY_SCOPE_AGENT);
    if (old == (unsigned)(NWG - 1)) {
      __hip_atomic_store(&bar[0], 0u, __ATOMIC_RELAXED, __HIP_MEMORY_SCOPE_AGENT);
      __hip_atomic_fetch_add(&bar[1], 1u, __ATOMIC_RELEASE, __HIP_MEMORY_SCOPE_AGENT);
    } else {
      while (__hip_atomic_load(&bar[1], __ATOMIC_ACQUIRE, __HIP_MEMORY_SCOPE_AGENT) == gen) {
#ifdef __HIP_DEVICE_COMPILE__
        __builtin_amdgcn_s_sleep(2);
#endif
      }
    }
  }
  __syncthreads();
}

// -------------------- prep kernels ----------------------------------------
// Zero h ping-pong buffers and barrier counters (must run before lstm_kernel).
__global__ void init_kernel(unsigned short* hbuf, unsigned* bar) {
  size_t i = (size_t)blockIdx.x * blockDim.x + threadIdx.x;
  if (i < 2u * BB * HH) hbuf[i] = 0;
  if (i == 0) { bar[0] = 0; bar[1] = 0; }
}

// prediction[:,0,:] = 0
__global__ void zeroslot_kernel(float* out) {
  size_t stride = (size_t)gridDim.x * blockDim.x;
  for (size_t i = (size_t)blockIdx.x * blockDim.x + threadIdx.x;
       i < (size_t)BB * VV; i += stride) {
    size_t b = i / VV, v = i % VV;
    out[b * ((size_t)TT * VV) + v] = 0.f;
  }
}

// Gather embeddings to bf16: out[(s*32+b)*128+e] = emb[idx[b*rowlen+s]][e]
__global__ void embed_kernel(const int* idx, const float* emb,
                             unsigned short* out, int steps, int rowlen) {
  size_t total  = (size_t)steps * BB * EE;
  size_t stride = (size_t)gridDim.x * blockDim.x;
  for (size_t i = (size_t)blockIdx.x * blockDim.x + threadIdx.x; i < total; i += stride) {
    int e = (int)(i & (EE - 1));
    int b = (int)((i >> 7) & (BB - 1));
    int s = (int)(i >> 12);
    int tok = idx[b * rowlen + s];
    out[i] = f2bf(emb[(size_t)tok * EE + e]);
  }
}

// Pack [Wih | Whh] (bf16) into per-WG WMMA B-fragment order:
// wpack[(((phase*16+wg)*4+ni)*12+kc)*512 + lane*16 + j]
//   N = wg*16 + lane%16 ; gate row r = ni*256 + N
//   K = kc*32 + (lane/16)*16 + j ; K<128 -> Wih[r][K] else Whh[r][K-128]
__global__ void pack_kernel(const float* WihE, const float* WhhE,
                            const float* WihD, const float* WhhD,
                            unsigned short* wpack) {
  size_t total  = (size_t)2 * NWG * 4 * KC12 * 32 * 16;  // 786432
  size_t stride = (size_t)gridDim.x * blockDim.x;
  for (size_t i = (size_t)blockIdx.x * blockDim.x + threadIdx.x; i < total; i += stride) {
    int j    = (int)(i & 15);
    int lane = (int)((i >> 4) & 31);
    size_t f = i >> 9;
    int kc = (int)(f % KC12); f /= KC12;
    int ni = (int)(f & 3);    f >>= 2;
    int wg = (int)(f & 15);   f >>= 4;
    int phase = (int)f;
    int n   = wg * 16 + (lane & 15);
    int row = ni * 256 + n;
    int K   = kc * 32 + ((lane >> 4) & 1) * 16 + j;
    const float* Wih = phase ? WihD : WihE;
    const float* Whh = phase ? WhhD : WhhE;
    float v = (K < EE) ? Wih[(size_t)row * EE + K]
                       : Whh[(size_t)row * HH + (K - EE)];
    wpack[i] = f2bf(v);
  }
}

__global__ void cvt_kernel(const float* src, unsigned short* dst, size_t n) {
  size_t stride = (size_t)gridDim.x * blockDim.x;
  for (size_t i = (size_t)blockIdx.x * blockDim.x + threadIdx.x; i < n; i += stride)
    dst[i] = f2bf(src[i]);
}

// -------------------- persistent LSTM kernel ------------------------------
// 16 WGs x 256 threads (8 waves). WG `wg` owns hidden columns [wg*16, wg*16+16)
// of all four gates (64 gate rows). Weights (48KB bf16) staged to LDS via TDM.
// Per step: g[32,64] = [x_t | h] @ Wslice^T via 8 waves x (1 C-tile, K=12 WMMAs),
// then gate nonlinearity + c/h update; h broadcast via global ping-pong buffer;
// steps separated by an agent-scope grid barrier.
__global__ __launch_bounds__(256, 1)
void lstm_kernel(const unsigned short* __restrict__ xe,
                 const unsigned short* __restrict__ xd,
                 const unsigned short* __restrict__ wpack,
                 const float* __restrict__ bE, const float* __restrict__ bD,
                 unsigned short* hbuf, unsigned short* hs,
                 float* hn_out, float* cn_out, unsigned* bar) {
  __shared__ unsigned short wlds[4 * KC12 * 512];  // 48KB: [ni][kc][lane*16+j]
  __shared__ float glds[2][4][16][16];             // [mi][gate][m][n]
  __shared__ float cstate[BB][16];

  const int tid  = threadIdx.x;
  const int wg   = blockIdx.x;      // 0..15
  const int lane = tid & 31;
  const int wv   = tid >> 5;        // 0..7
  const int mi   = wv & 1;          // batch halves
  const int ni   = wv >> 1;         // gate (i,f,g,o)
  const int lrow = lane & 15;
  const int kh   = (lane >> 4) & 1;

  for (int e = tid; e < BB * 16; e += 256) cstate[e >> 4][e & 15] = 0.f;

  int cur = 0;
  for (int phase = 0; phase < 2; ++phase) {
    // ---- stage this phase's weight slice into LDS ----
#if defined(__HIP_DEVICE_COMPILE__) && __has_builtin(__builtin_amdgcn_tensor_load_to_lds)
    if (wv == 0) {
      uint64_t ga  = (uint64_t)(uintptr_t)(wpack + (size_t)(phase * NWG + wg) * 24576u);
      unsigned lo  = (unsigned)ga;
      unsigned hi  = (unsigned)(ga >> 32);
      unsigned lad = (unsigned)(uintptr_t)(&wlds[0]);  // LDS offset in low 32 bits
      // D# group0: count=1, lds_addr, global_addr, type=2 (bits 127:126)
      u32x4 g0 = { 1u, lad, lo, (hi & 0x01FFFFFFu) | 0x80000000u };
      // D# group1: data_size=4B, tensor_dim0=tile_dim0=24576 dwords, dim1=tile1=1,
      // stride0=24576 (flat contiguous copy of 96KB as one row)
      i32x8 g1 = { (int)(2u << 16),
                   (int)(24576u << 16),   // dim0[15:0] in bits 63:48
                   (int)(1u << 16),       // dim0[31:16]=0 | dim1[15:0]=1
                   (int)(24576u << 16),   // dim1[31:16]=0 | tile_dim0
                   1,                     // tile_dim1=1 | tile_dim2=0
                   24576,                 // tensor_dim0_stride lo32
                   0, 0 };
      i32x4 z4 = { 0, 0, 0, 0 };
#if defined(__clang_major__) && (__clang_major__ >= 23)
      i32x8 z8 = { 0, 0, 0, 0, 0, 0, 0, 0 };
      __builtin_amdgcn_tensor_load_to_lds(g0, g1, z4, z4, z8, 0);
#else
      __builtin_amdgcn_tensor_load_to_lds(g0, g1, z4, z4, 0);
#endif
#if __has_builtin(__builtin_amdgcn_s_wait_tensorcnt)
      __builtin_amdgcn_s_wait_tensorcnt(0);
#endif
    }
#else
    {  // fallback: cooperative copy (3072 x b128)
      const uint4* src = (const uint4*)(wpack + (size_t)(phase * NWG + wg) * 24576u);
      uint4* dst = (uint4*)&wlds[0];
      for (int i = tid; i < 3072; i += 256) dst[i] = src[i];
    }
#endif
    __syncthreads();

    const int nsteps = phase ? (TT - 1) : SS;
    const unsigned short* xbase = phase ? xd : xe;
    const float* bias = phase ? bD : bE;

    for (int st = 0; st < nsteps; ++st) {
      const unsigned short* xt  = xbase + (size_t)st * (BB * EE);
      const unsigned short* hp  = hbuf + (size_t)cur * (BB * HH);
      unsigned short*       hnx = hbuf + (size_t)(cur ^ 1) * (BB * HH);

      // ---- WMMA: this wave's 16x16 gate tile, K = 384 ----
      v8f acc = { 0.f, 0.f, 0.f, 0.f, 0.f, 0.f, 0.f, 0.f };
#pragma unroll
      for (int kc = 0; kc < KC12; ++kc) {
        Frag a, bf;
        const unsigned short* ap =
            (kc < 4) ? (xt + (size_t)(mi * 16 + lrow) * EE + kc * 32)
                     : (hp + (size_t)(mi * 16 + lrow) * HH + (kc - 4) * 32);
        a.q[0] = *(const uint4*)(ap + kh * 16);
        a.q[1] = *(const uint4*)(ap + kh * 16 + 8);
        const unsigned short* bp = &wlds[((ni * KC12) + kc) * 512 + lane * 16];
        bf.q[0] = ((const uint4*)bp)[0];
        bf.q[1] = ((const uint4*)bp)[1];
        acc = wmma_bf16(a.b, bf.b, acc);
      }
#pragma unroll
      for (int r = 0; r < 8; ++r) glds[mi][ni][r + kh * 8][lrow] = acc[r];
      __syncthreads();

      // ---- gate combine + state update (512 (b,hc) pairs / 256 threads) ----
      const int last = (phase == 1) && (st == nsteps - 1);
      for (int e = tid; e < BB * 16; e += 256) {
        int b = e >> 4, hc = e & 15;
        int col = wg * 16 + hc;
        float gi = glds[b >> 4][0][b & 15][hc] + bias[0 * HH + col];
        float gf = glds[b >> 4][1][b & 15][hc] + bias[1 * HH + col];
        float gg = glds[b >> 4][2][b & 15][hc] + bias[2 * HH + col];
        float go = glds[b >> 4][3][b & 15][hc] + bias[3 * HH + col];
        float c = sigmoidf_(gf) * cstate[b][hc] + sigmoidf_(gi) * tanhf_(gg);
        float h = sigmoidf_(go) * tanhf_(c);
        cstate[b][hc] = c;
        unsigned short hb = f2bf(h);
        hnx[(size_t)b * HH + col] = hb;
        if (phase) hs[((size_t)st * BB + b) * HH + col] = hb;
        if (last) {
          hn_out[(size_t)b * HH + col] = h;
          cn_out[(size_t)b * HH + col] = c;
        }
      }
      cur ^= 1;
      grid_barrier(bar);
    }
  }
}

// -------------------- FC projection: bf16 WMMA GEMM -----------------------
// grid (125, 63), block 128 (4 waves). Wave w: N = bx*256 + w*64 (4 tiles),
// M = 32 (2 tiles, one decoder step t), K = 256 (8 chunks).
__global__ __launch_bounds__(128, 2)
void fc_kernel(const unsigned short* __restrict__ hsb,
               const unsigned short* __restrict__ wfb,
               const float* __restrict__ bfc, float* __restrict__ out) {
  const int t    = blockIdx.y;            // 0..62
  const int wv   = threadIdx.x >> 5;      // 0..3
  const int lane = threadIdx.x & 31;
  const int lrow = lane & 15;
  const int kh   = (lane >> 4) & 1;
  const int n0   = blockIdx.x * 256 + wv * 64;

  const unsigned short* A = hsb + (size_t)t * (BB * HH);

  v8f acc[2][4];
#pragma unroll
  for (int m = 0; m < 2; ++m)
#pragma unroll
    for (int nn = 0; nn < 4; ++nn)
      acc[m][nn] = v8f{ 0.f, 0.f, 0.f, 0.f, 0.f, 0.f, 0.f, 0.f };

#pragma unroll
  for (int kc = 0; kc < 8; ++kc) {
    Frag a[2], bb[4];
#pragma unroll
    for (int m = 0; m < 2; ++m) {
      const unsigned short* ap = A + (size_t)(m * 16 + lrow) * HH + kc * 32;
      a[m].q[0] = *(const uint4*)(ap + kh * 16);
      a[m].q[1] = *(const uint4*)(ap + kh * 16 + 8);
    }
#pragma unroll
    for (int nn = 0; nn < 4; ++nn) {
      const unsigned short* bp =
          wfb + (size_t)(n0 + nn * 16 + lrow) * HH + kc * 32 + kh * 16;
      bb[nn].q[0] = ((const uint4*)bp)[0];
      bb[nn].q[1] = ((const uint4*)bp)[1];
    }
#pragma unroll
    for (int m = 0; m < 2; ++m)
#pragma unroll
      for (int nn = 0; nn < 4; ++nn)
        acc[m][nn] = wmma_bf16(a[m].b, bb[nn].b, acc[m][nn]);
  }

#pragma unroll
  for (int nn = 0; nn < 4; ++nn) {
    const int n = n0 + nn * 16 + lrow;
    const float bias = bfc[n];
#pragma unroll
    for (int m = 0; m < 2; ++m) {
      const int brow = m * 16 + kh * 8;
#pragma unroll
      for (int r = 0; r < 8; ++r) {
        int b = brow + r;
        out[((size_t)b * TT + (t + 1)) * VV + n] = acc[m][nn][r] + bias;
      }
    }
  }
}

// -------------------- host entry ------------------------------------------
extern "C" void kernel_launch(void* const* d_in, const int* in_sizes, int n_in,
                              void* d_out, int out_size, void* d_ws, size_t ws_size,
                              hipStream_t stream) {
  const int*   src  = (const int*)d_in[0];
  const int*   trg  = (const int*)d_in[1];
  const float* embS = (const float*)d_in[2];
  const float* WihE = (const float*)d_in[3];
  const float* WhhE = (const float*)d_in[4];
  const float* bE   = (const float*)d_in[5];
  const float* embT = (const float*)d_in[6];
  const float* WihD = (const float*)d_in[7];
  const float* WhhD = (const float*)d_in[8];
  const float* bD   = (const float*)d_in[9];
  const float* Wfc  = (const float*)d_in[10];
  const float* bfc  = (const float*)d_in[11];
  float* out = (float*)d_out;

  // workspace layout (bytes, 256-aligned)
  char* ws = (char*)d_ws;
  unsigned short* xe   = (unsigned short*)(ws + 0);          // 128*32*128 bf16 = 1,048,576
  unsigned short* xd   = (unsigned short*)(ws + 1048576);    // 63*32*128 bf16  =   516,096
  unsigned short* wpk  = (unsigned short*)(ws + 1564672);    // 2*16*24576 u16  = 1,572,864
  unsigned short* wfb  = (unsigned short*)(ws + 3137536);    // 32000*256 bf16  = 16,384,000
  unsigned short* hs   = (unsigned short*)(ws + 19521536);   // 63*32*256 bf16  = 1,032,192
  unsigned short* hbuf = (unsigned short*)(ws + 20553728);   // 2*32*256 bf16   =    32,768
  unsigned*       bar  = (unsigned*)(ws + 20586496);         // barrier counters

  init_kernel<<<64, 256, 0, stream>>>(hbuf, bar);
  embed_kernel<<<2048, 256, 0, stream>>>(src, embS, xe, SS, SS);
  embed_kernel<<<1008, 256, 0, stream>>>(trg, embT, xd, TT - 1, TT);
  pack_kernel<<<1536, 256, 0, stream>>>(WihE, WhhE, WihD, WhhD, wpk);
  cvt_kernel<<<4096, 256, 0, stream>>>(Wfc, wfb, (size_t)VV * HH);
  zeroslot_kernel<<<2000, 256, 0, stream>>>(out);

  float* hn_out = out + (size_t)BB * TT * VV;
  float* cn_out = hn_out + (size_t)BB * HH;
  lstm_kernel<<<NWG, 256, 0, stream>>>(xe, xd, wpk, bE, bD, hbuf, hs,
                                       hn_out, cn_out, bar);
  fc_kernel<<<dim3(125, 63), 128, 0, stream>>>(hs, wfb, bfc, out);

  (void)in_sizes; (void)n_in; (void)out_size; (void)ws_size;
}
